// BilinearAttention_31576599560680
// MI455X (gfx1250) — matrix operand order repaired and checked by reading, not compiled, not run on hardware
//
#include <hip/hip_runtime.h>

typedef __attribute__((ext_vector_type(16))) _Float16 v16h;
typedef __attribute__((ext_vector_type(8)))  float    v8f;

#define NEG_INF (-1e30f)

constexpr int MTOK   = 200;   // tokens per batch row
constexpr int MPAD   = 208;   // 13 tiles of 16
constexpr int NTILES = 13;
constexpr int DQ     = 512;
constexpr int DT     = 768;
constexpr int DA     = 64;
constexpr int KSTEP  = 32;
constexpr int NWAVES = 8;

union V16HU { v16h v; _Float16 h[16]; };

__device__ inline void cvt8(_Float16* dst, float4 a, float4 b) {
    dst[0] = (_Float16)a.x; dst[1] = (_Float16)a.y;
    dst[2] = (_Float16)a.z; dst[3] = (_Float16)a.w;
    dst[4] = (_Float16)b.x; dst[5] = (_Float16)b.y;
    dst[6] = (_Float16)b.z; dst[7] = (_Float16)b.w;
}

// One-time: transpose + convert Wk/Wv (768x64 f32, row-major) into
// WkT/WvT (64x768 f16) so B operands are contiguous b128 loads from L2.
__global__ __launch_bounds__(256)
void prep_weights(const float* __restrict__ Wk, const float* __restrict__ Wv,
                  _Float16* __restrict__ wkT, _Float16* __restrict__ wvT)
{
    int idx = blockIdx.x * 256 + threadIdx.x;   // 0 .. 64*768-1
    int n = idx / DT;
    int k = idx % DT;
    wkT[idx] = (_Float16)Wk[k * DA + n];
    wvT[idx] = (_Float16)Wv[k * DA + n];
}

__global__ __launch_bounds__(256)
void bilinear_attn_fused(const float* __restrict__ q_vec,
                         const float* __restrict__ desc,
                         const int*   __restrict__ dmask,
                         const float* __restrict__ Wq,
                         const float* __restrict__ U,
                         const float* __restrict__ Wo,
                         const float* __restrict__ bo,
                         const _Float16* __restrict__ wkT,
                         const _Float16* __restrict__ wvT,
                         float* __restrict__ out_fused,
                         float* __restrict__ out_attn)
{
    __shared__ float    s_q[DQ];              // q_vec row (reused at the end)
    __shared__ float    s_Q[DA];
    __shared__ float    s_QU[DA];
    __shared__ _Float16 s_V[MPAD * DA];       // V rows, f16
    __shared__ float    s_score[MPAD];
    __shared__ float    s_attn[MPAD];
    __shared__ float    s_ctxp[4][DA];
    __shared__ float    s_ctx[DA];
    __shared__ float    s_red[2];

    const int b    = blockIdx.x;
    const int tid  = threadIdx.x;
    const int lane = tid & 31;
    const int wave = tid >> 5;
    const int hsel = (lane >> 4) & 1;   // which 16-lane half
    const int l15  = lane & 15;

    // ---------- QU = (q_vec[b] @ Wq) @ U  (tiny, VALU) ----------
    for (int i = tid; i < DQ; i += 256) s_q[i] = q_vec[(size_t)b * DQ + i];
    __syncthreads();
    if (tid < DA) {
        float acc = 0.f;
        for (int t = 0; t < DQ; ++t) acc += s_q[t] * Wq[t * DA + tid];
        s_Q[tid] = acc;
    }
    __syncthreads();
    if (tid < DA) {
        float acc = 0.f;
        for (int d = 0; d < DA; ++d) acc += s_Q[d] * U[d * DA + tid];
        s_QU[tid] = acc;
    }
    __syncthreads();

    // ---------- K/V projection with WMMA, fused scores (barrier-free loop) ----------
    for (int pass = 0; pass < 2; ++pass) {
        const int tile = wave + pass * NWAVES;          // 0..15 (13 real)
        // A-operand row for this lane (16-bit A 16x32 layout: lane = M)
        int row  = tile * 16 + l15;
        int rowc = row < MTOK ? row : MTOK - 1;         // branchless clamp, EXEC stays full
        const float* arow = desc + ((size_t)b * MTOK + rowc) * DT;

        v8f zero = {0.f,0.f,0.f,0.f,0.f,0.f,0.f,0.f};
        v8f accK[4], accV[4];
        #pragma unroll
        for (int nt = 0; nt < 4; ++nt) { accK[nt] = zero; accV[nt] = zero; }

        const int kbase = hsel * 16;   // B: lane<16 holds k 0..15, lane>=16 holds 16..31
        for (int k0 = 0; k0 < DT; k0 += KSTEP) {
            // A operand: lane<16 -> K {0..15}; lane>=16 -> {16..31}
            V16HU a;
            {
                int cbase = k0 + hsel * 16;
                const float4* ap = (const float4*)(arow + cbase);
                float4 f0 = ap[0], f1 = ap[1], f2 = ap[2], f3 = ap[3];
                cvt8(&a.h[0], f0, f1);
                cvt8(&a.h[8], f2, f3);
            }
            if (k0 + KSTEP < DT)   // hide HBM latency for the next A slab
                __builtin_prefetch(arow + k0 + KSTEP + hsel * 16, 0, 0);

            #pragma unroll
            for (int nt = 0; nt < 4; ++nt) {
                int n = nt * 16 + l15;
                v16h bk = *(const v16h*)(wkT + (size_t)n * DT + k0 + kbase);
                v16h bv = *(const v16h*)(wvT + (size_t)n * DT + k0 + kbase);
                accK[nt] = __builtin_amdgcn_wmma_f32_16x16x32_f16(
                    false, a.v, false, bk, (short)0, accK[nt], false, false);
                accV[nt] = __builtin_amdgcn_wmma_f32_16x16x32_f16(
                    false, a.v, false, bv, (short)0, accV[nt], false, false);
            }
        }

        // scores s[m] = QU . K[m,:] straight from the C layout:
        // element (r,lane): M = r + 8*hsel, N = nt*16 + l15
        float p[8];
        #pragma unroll
        for (int r = 0; r < 8; ++r) p[r] = 0.f;
        #pragma unroll
        for (int nt = 0; nt < 4; ++nt) {
            float qu = s_QU[nt * 16 + l15];
            #pragma unroll
            for (int r = 0; r < 8; ++r) p[r] += qu * accK[nt][r];
        }
        #pragma unroll
        for (int r = 0; r < 8; ++r) {   // reduce within each 16-lane half
            p[r] += __shfl_xor(p[r], 1);
            p[r] += __shfl_xor(p[r], 2);
            p[r] += __shfl_xor(p[r], 4);
            p[r] += __shfl_xor(p[r], 8);
        }
        if (tile < NTILES) {            // wave-uniform guard
            if (l15 == 0) {
                int mbase = tile * 16 + hsel * 8;
                #pragma unroll
                for (int r = 0; r < 8; ++r) s_score[mbase + r] = p[r];
            }
            #pragma unroll
            for (int nt = 0; nt < 4; ++nt) {
                int d = nt * 16 + l15;
                #pragma unroll
                for (int r = 0; r < 8; ++r) {
                    int m = tile * 16 + hsel * 8 + r;
                    s_V[m * DA + d] = (_Float16)accV[nt][r];
                }
            }
        }
    }
    __syncthreads();

    // ---------- masked softmax over M ----------
    if (tid < MPAD) {
        float s = s_score[tid];
        bool valid = (tid < MTOK) && (dmask[(size_t)b * MTOK + tid] != 0);
        s_score[tid] = valid ? s : NEG_INF;
    }
    __syncthreads();
    if (wave == 0) {
        float mx = NEG_INF;
        for (int m = lane; m < MPAD; m += 32) mx = fmaxf(mx, s_score[m]);
        mx = fmaxf(mx, __shfl_xor(mx, 1));
        mx = fmaxf(mx, __shfl_xor(mx, 2));
        mx = fmaxf(mx, __shfl_xor(mx, 4));
        mx = fmaxf(mx, __shfl_xor(mx, 8));
        mx = fmaxf(mx, __shfl_xor(mx, 16));
        float sum = 0.f;
        for (int m = lane; m < MPAD; m += 32) sum += __expf(s_score[m] - mx);
        sum += __shfl_xor(sum, 1);
        sum += __shfl_xor(sum, 2);
        sum += __shfl_xor(sum, 4);
        sum += __shfl_xor(sum, 8);
        sum += __shfl_xor(sum, 16);
        if (lane == 0) { s_red[0] = mx; s_red[1] = 1.f / sum; }
    }
    __syncthreads();
    if (tid < MPAD) {
        float a = __expf(s_score[tid] - s_red[0]) * s_red[1];
        s_attn[tid] = a;
        if (tid < MTOK) out_attn[(size_t)b * MTOK + tid] = a;
    }
    __syncthreads();

    // ---------- context = attn @ V ----------
    {
        int d = tid & 63;
        int q4 = tid >> 6;
        float acc = 0.f;
        for (int m = q4; m < MPAD; m += 4)
            acc += s_attn[m] * (float)s_V[m * DA + d];
        s_ctxp[q4][d] = acc;
    }
    __syncthreads();
    if (tid < DA)
        s_ctx[tid] = s_ctxp[0][tid] + s_ctxp[1][tid] + s_ctxp[2][tid] + s_ctxp[3][tid];
    __syncthreads();

    // ---------- fused = context @ Wo + bo + q_vec ----------
    for (int j = tid; j < DQ; j += 256) {
        float acc = bo[j] + s_q[j];
        for (int d = 0; d < DA; ++d) acc += s_ctx[d] * Wo[d * DQ + j];
        out_fused[(size_t)b * DQ + j] = acc;
    }
}

extern "C" void kernel_launch(void* const* d_in, const int* in_sizes, int n_in,
                              void* d_out, int out_size, void* d_ws, size_t ws_size,
                              hipStream_t stream) {
    const float* q_vec = (const float*)d_in[0];
    const float* desc  = (const float*)d_in[1];
    const int*   dmask = (const int*)d_in[2];
    const float* Wq    = (const float*)d_in[3];
    const float* Wk    = (const float*)d_in[4];
    const float* Wv    = (const float*)d_in[5];
    const float* U     = (const float*)d_in[6];
    const float* Wo    = (const float*)d_in[7];
    const float* bo    = (const float*)d_in[8];

    const int B = in_sizes[0] / DQ;     // 2048
    float* out_fused = (float*)d_out;                  // (B, 512)
    float* out_attn  = out_fused + (size_t)B * DQ;     // (B, 200)

    _Float16* wkT = (_Float16*)d_ws;                   // 64x768 f16
    _Float16* wvT = wkT + (size_t)DA * DT;             // 64x768 f16

    prep_weights<<<(DA * DT) / 256, 256, 0, stream>>>(Wk, Wv, wkT, wvT);
    bilinear_attn_fused<<<B, 256, 0, stream>>>(
        q_vec, desc, dmask, Wq, U, Wo, bo, wkT, wvT, out_fused, out_attn);
}
